// NsNet2_npy_77249281786020
// MI455X (gfx1250) — matrix operand order repaired and checked by reading, not compiled, not run on hardware
//
#include <hip/hip_runtime.h>
#include <hip/hip_bf16.h>
#include <stdint.h>

typedef __bf16 bf16_t;
typedef __attribute__((ext_vector_type(16))) __bf16 v16bf;
typedef __attribute__((ext_vector_type(8)))  float  v8f;

// Problem dims
constexpr int Bsz = 32768, Fdim = 257, Hdim = 400, FFdim = 600;
// Padded K dims (multiples of 32)
constexpr int KP_F  = 288;   // pad of 257
constexpr int KP_H  = 416;   // pad of 400
constexpr int KP_FF = 608;   // pad of 600
// N rounded to block tile (128)
constexpr int NR_H  = 512;   // 400 -> 512
constexpr int NR_FF = 640;   // 600 -> 640
constexpr int NR_F  = 384;   // 257 -> 384
constexpr int ACT_LD = 640;  // activation buffer stride (bf16)

union FragU { uint4 u[2]; v16bf v; };

__device__ __forceinline__ v8f wmma_bf16(const v16bf& a, const v16bf& b, const v8f& c)
{
    return __builtin_amdgcn_wmma_f32_16x16x32_bf16(false, a, false, b, (short)0, c, false, false);
}

__device__ __forceinline__ float sigmoidf_(float x) { return 1.f / (1.f + __expf(-x)); }

// ---------------------------------------------------------------------------
// Generic bf16 WMMA GEMM:  C[M x N] = A[M x Kp] * W^T  (W packed [Nround x Kp])
// Block tile 128x128, 8 waves, wave tile 32x64 (2x4 WMMA 16x16 frags).
// mode 1: bias, bf16 store (0 for n>=N)    -> Cb, ldo
// mode 2: bias+relu, bf16 store            -> Cb, ldo
// mode 3: bias+sigmoid, fp32 (guard n<N)   -> Cf, ldc
// ---------------------------------------------------------------------------
__global__ __launch_bounds__(256)
void gemm_bf16_wmma(const bf16_t* __restrict__ A, int lda,
                    const bf16_t* __restrict__ Wt, int ldb,
                    const float* __restrict__ bias,
                    float* __restrict__ Cf, int ldc,
                    bf16_t* __restrict__ Cb, int ldo,
                    int N, int Kp, int mode)
{
    const int lane = threadIdx.x & 31;
    const int wave = threadIdx.x >> 5;
    const int lr   = lane & 15;
    const int half = lane >> 4;                 // 0: lanes 0-15, 1: lanes 16-31
    const int bm = blockIdx.y * 128 + (wave & 3) * 32;
    const int bn = blockIdx.x * 128 + (wave >> 2) * 64;

    v8f acc[2][4] = {};

    const bf16_t* a0p = A + (size_t)(bm + lr) * lda + half * 8;
    const bf16_t* a1p = a0p + (size_t)16 * lda;

    for (int k0 = 0; k0 < Kp; k0 += 32) {
        FragU a0, a1;
        a0.u[0] = *(const uint4*)(a0p + k0);
        a0.u[1] = *(const uint4*)(a0p + k0 + 16);
        a1.u[0] = *(const uint4*)(a1p + k0);
        a1.u[1] = *(const uint4*)(a1p + k0 + 16);

        #pragma unroll
        for (int j = 0; j < 4; ++j) {
            FragU b;
            const bf16_t* bp = Wt + (size_t)(bn + j * 16 + lr) * ldb + k0 + half * 8;
            b.u[0] = *(const uint4*)(bp);
            b.u[1] = *(const uint4*)(bp + 16);
            acc[0][j] = wmma_bf16(a0.v, b.v, acc[0][j]);
            acc[1][j] = wmma_bf16(a1.v, b.v, acc[1][j]);
        }
    }

    #pragma unroll
    for (int i = 0; i < 2; ++i) {
        #pragma unroll
        for (int j = 0; j < 4; ++j) {
            const int n = bn + j * 16 + lr;
            #pragma unroll
            for (int r = 0; r < 8; ++r) {
                const int m = bm + i * 16 + half * 8 + r;
                const float v = acc[i][j][r];
                if (mode == 3) {
                    if (n < N) Cf[(size_t)m * ldc + n] = sigmoidf_(v + bias[n]);
                } else {
                    float o = 0.f;
                    if (n < N) {
                        o = v + bias[n];
                        if (mode == 2 && o < 0.f) o = 0.f;
                    }
                    Cb[(size_t)m * ldo + n] = (bf16_t)o;
                }
            }
        }
    }
}

// ---------------------------------------------------------------------------
// Fully fused GRU cell:
//   z = sig(xi@Wiz^T + h@Whz^T + biz + bhz)
//   r = sig(xi@Wir^T + h@Whr^T + bir + bhr)
//   n = tanh(xi@Win^T + bin + r*(h@Whn^T + bhn))
//   h' = (1-z)*n + z*h           -> written as bf16 (zero-padded to ACT_LD)
// Four accumulator sets per wave (z, r summed across both inputs; nx, nh kept
// separate). Weights packed per-gate: Wi/Wh = 3 x [512 x 416] bf16.
// Block tile 128x128 over (M, gate-col), 8 waves, wave tile 32x64.
// ---------------------------------------------------------------------------
__global__ __launch_bounds__(256)
void gru_fused_wmma(const bf16_t* __restrict__ Axi,   // [B x ACT_LD]
                    const bf16_t* __restrict__ Ah,    // [B x ACT_LD] (bf16 h)
                    const bf16_t* __restrict__ Wi,    // 3 x [512 x KP_H]
                    const bf16_t* __restrict__ Wh,    // 3 x [512 x KP_H]
                    const float* __restrict__ hf,     // [B x Hdim] fp32 h
                    const float* __restrict__ biz, const float* __restrict__ bir,
                    const float* __restrict__ bin_,
                    const float* __restrict__ bhz, const float* __restrict__ bhr,
                    const float* __restrict__ bhn,
                    bf16_t* __restrict__ g)           // [B x ACT_LD]
{
    const int lane = threadIdx.x & 31;
    const int wave = threadIdx.x >> 5;
    const int lr   = lane & 15;
    const int half = lane >> 4;
    const int bm = blockIdx.y * 128 + (wave & 3) * 32;
    const int bn = blockIdx.x * 128 + (wave >> 2) * 64;

    constexpr size_t GATE_W = (size_t)512 * KP_H;

    v8f accZ[2][4] = {}, accR[2][4] = {}, accNX[2][4] = {}, accNH[2][4] = {};

    const bf16_t* ax0 = Axi + (size_t)(bm + lr) * ACT_LD + half * 8;
    const bf16_t* ax1 = ax0 + (size_t)16 * ACT_LD;
    const bf16_t* ah0 = Ah  + (size_t)(bm + lr) * ACT_LD + half * 8;
    const bf16_t* ah1 = ah0 + (size_t)16 * ACT_LD;

    for (int k0 = 0; k0 < KP_H; k0 += 32) {
        FragU axi0, axi1, ahh0, ahh1;
        axi0.u[0] = *(const uint4*)(ax0 + k0);
        axi0.u[1] = *(const uint4*)(ax0 + k0 + 16);
        axi1.u[0] = *(const uint4*)(ax1 + k0);
        axi1.u[1] = *(const uint4*)(ax1 + k0 + 16);
        ahh0.u[0] = *(const uint4*)(ah0 + k0);
        ahh0.u[1] = *(const uint4*)(ah0 + k0 + 16);
        ahh1.u[0] = *(const uint4*)(ah1 + k0);
        ahh1.u[1] = *(const uint4*)(ah1 + k0 + 16);

        auto mmaW = [&](const bf16_t* Wbase, const FragU& A0, const FragU& A1,
                        v8f (&acc)[2][4]) {
            #pragma unroll
            for (int j = 0; j < 4; ++j) {
                FragU b;
                const bf16_t* bp = Wbase + (size_t)(bn + j * 16 + lr) * KP_H + k0 + half * 8;
                b.u[0] = *(const uint4*)(bp);
                b.u[1] = *(const uint4*)(bp + 16);
                acc[0][j] = wmma_bf16(A0.v, b.v, acc[0][j]);
                acc[1][j] = wmma_bf16(A1.v, b.v, acc[1][j]);
            }
        };

        mmaW(Wi + 0 * GATE_W, axi0, axi1, accZ);   // z: xi side
        mmaW(Wh + 0 * GATE_W, ahh0, ahh1, accZ);   // z: h side (summed)
        mmaW(Wi + 1 * GATE_W, axi0, axi1, accR);   // r: xi side
        mmaW(Wh + 1 * GATE_W, ahh0, ahh1, accR);   // r: h side (summed)
        mmaW(Wi + 2 * GATE_W, axi0, axi1, accNX);  // n: xi side (separate)
        mmaW(Wh + 2 * GATE_W, ahh0, ahh1, accNH);  // n: h side (separate)
    }

    // Register-level GRU epilogue; C layout: col n = bn+j*16+lr, row m = bm+i*16+half*8+r
    #pragma unroll
    for (int i = 0; i < 2; ++i) {
        #pragma unroll
        for (int j = 0; j < 4; ++j) {
            const int n = bn + j * 16 + lr;
            float bz = 0.f, br = 0.f, bn2 = 0.f, bhn2 = 0.f;
            if (n < Hdim) {
                bz = biz[n] + bhz[n];
                br = bir[n] + bhr[n];
                bn2 = bin_[n];
                bhn2 = bhn[n];
            }
            #pragma unroll
            for (int r = 0; r < 8; ++r) {
                const int m = bm + i * 16 + half * 8 + r;
                float o = 0.f;
                if (n < Hdim) {
                    const float z  = sigmoidf_(accZ[i][j][r] + bz);
                    const float rr = sigmoidf_(accR[i][j][r] + br);
                    const float nn = tanhf(accNX[i][j][r] + bn2 + rr * (accNH[i][j][r] + bhn2));
                    const float hv = hf[(size_t)m * Hdim + n];
                    o = (1.f - z) * nn + z * hv;
                }
                g[(size_t)m * ACT_LD + n] = (bf16_t)o;
            }
        }
    }
}

// fp32 -> zero-padded bf16 activation buffer
__global__ void cvt_pad(const float* __restrict__ src, int scols,
                        bf16_t* __restrict__ dst, int dcols)
{
    const int c = blockIdx.x * blockDim.x + threadIdx.x;
    const int b = blockIdx.y;
    if (c >= dcols) return;
    dst[(size_t)b * dcols + c] = (c < scols) ? (bf16_t)src[(size_t)b * scols + c] : (bf16_t)0.f;
}

// fp32 [N x K] row-major -> bf16 padded [.. x Kp] (arena pre-zeroed)
__global__ void pack_w(const float* __restrict__ src, bf16_t* __restrict__ dst,
                       int N, int K, int Kp)
{
    const int i = blockIdx.x * blockDim.x + threadIdx.x;
    if (i >= N * K) return;
    const int n = i / K, k = i % K;
    dst[(size_t)n * Kp + k] = (bf16_t)src[i];
}

__global__ void fill0(uint32_t* __restrict__ p, long n)
{
    const long i = (long)blockIdx.x * blockDim.x + threadIdx.x;
    if (i < n) p[i] = 0u;
}

// Weight arena offsets (bf16 elements); GRU weights are per-gate [512 x KP_H]
constexpr size_t GATE_W = (size_t)512 * KP_H;
constexpr size_t OF_FC1 = 0;                               // 512 x 288
constexpr size_t OF_GI1 = OF_FC1 + (size_t)NR_H * KP_F;    // 3 gates
constexpr size_t OF_GH1 = OF_GI1 + 3 * GATE_W;
constexpr size_t OF_GI2 = OF_GH1 + 3 * GATE_W;
constexpr size_t OF_GH2 = OF_GI2 + 3 * GATE_W;
constexpr size_t OF_FC2 = OF_GH2 + 3 * GATE_W;             // 640 x 416
constexpr size_t OF_FC3 = OF_FC2 + (size_t)NR_FF * KP_H;   // 640 x 608
constexpr size_t OF_FC4 = OF_FC3 + (size_t)NR_FF * KP_FF;  // 384 x 608
constexpr size_t W_ELEMS = OF_FC4 + (size_t)NR_F * KP_FF;

extern "C" void kernel_launch(void* const* d_in, const int* in_sizes, int n_in,
                              void* d_out, int out_size, void* d_ws, size_t ws_size,
                              hipStream_t stream)
{
    (void)in_sizes; (void)n_in; (void)out_size; (void)ws_size;

    const float* x    = (const float*)d_in[0];
    const float* h1   = (const float*)d_in[1];
    const float* h2   = (const float*)d_in[2];
    float* out        = (float*)d_out;

    uint8_t* ws = (uint8_t*)d_ws;
    size_t o = 0;
    auto alloc = [&](size_t bytes) -> size_t {
        size_t r = o;
        o = (o + bytes + 255) & ~(size_t)255;
        return r;
    };
    const size_t oW    = alloc(W_ELEMS * sizeof(bf16_t));
    const size_t oXbf  = alloc((size_t)Bsz * KP_F * sizeof(bf16_t));
    const size_t oAct0 = alloc((size_t)Bsz * ACT_LD * sizeof(bf16_t));
    const size_t oAct1 = alloc((size_t)Bsz * ACT_LD * sizeof(bf16_t));
    const size_t oHbf  = alloc((size_t)Bsz * ACT_LD * sizeof(bf16_t));

    bf16_t* warena = (bf16_t*)(ws + oW);
    bf16_t* xbf    = (bf16_t*)(ws + oXbf);
    bf16_t* act0   = (bf16_t*)(ws + oAct0);
    bf16_t* act1   = (bf16_t*)(ws + oAct1);
    bf16_t* hbf    = (bf16_t*)(ws + oHbf);

    // 1) zero weight arena (padding must be 0), then pack weights to bf16
    {
        const long words = (long)((W_ELEMS * sizeof(bf16_t) + 3) / 4);
        fill0<<<dim3((unsigned)((words + 255) / 256)), dim3(256), 0, stream>>>((uint32_t*)warena, words);
    }
    auto P = [&](int idx, size_t dstOfs, int N, int K, int Kp) {
        const int tot = N * K;
        pack_w<<<dim3((tot + 255) / 256), dim3(256), 0, stream>>>(
            (const float*)d_in[idx], warena + dstOfs, N, K, Kp);
    };
    P(3,  OF_FC1, Hdim, Fdim, KP_F);                 // Wfc1
    // GRU layer 1 (gate order z, r, n)
    P(5,  OF_GI1 + 0 * GATE_W, Hdim, Hdim, KP_H);    // Wiz1
    P(6,  OF_GI1 + 1 * GATE_W, Hdim, Hdim, KP_H);    // Wir1
    P(7,  OF_GI1 + 2 * GATE_W, Hdim, Hdim, KP_H);    // Win1
    P(8,  OF_GH1 + 0 * GATE_W, Hdim, Hdim, KP_H);    // Whz1
    P(9,  OF_GH1 + 1 * GATE_W, Hdim, Hdim, KP_H);    // Whr1
    P(10, OF_GH1 + 2 * GATE_W, Hdim, Hdim, KP_H);    // Whn1
    // GRU layer 2
    P(17, OF_GI2 + 0 * GATE_W, Hdim, Hdim, KP_H);
    P(18, OF_GI2 + 1 * GATE_W, Hdim, Hdim, KP_H);
    P(19, OF_GI2 + 2 * GATE_W, Hdim, Hdim, KP_H);
    P(20, OF_GH2 + 0 * GATE_W, Hdim, Hdim, KP_H);
    P(21, OF_GH2 + 1 * GATE_W, Hdim, Hdim, KP_H);
    P(22, OF_GH2 + 2 * GATE_W, Hdim, Hdim, KP_H);
    P(29, OF_FC2, FFdim, Hdim,  KP_H);               // Wfc2
    P(31, OF_FC3, FFdim, FFdim, KP_FF);              // Wfc3
    P(33, OF_FC4, Fdim,  FFdim, KP_FF);              // Wfc4

    // 2) convert inputs to padded bf16
    cvt_pad<<<dim3((KP_F + 255) / 256, Bsz), dim3(256), 0, stream>>>(x, Fdim, xbf, KP_F);
    cvt_pad<<<dim3((ACT_LD + 255) / 256, Bsz), dim3(256), 0, stream>>>(h1, Hdim, hbf, ACT_LD);

    auto G = [&](const bf16_t* A, int lda, size_t wOfs, int ldb, const float* bias,
                 float* Cf, int ldc, bf16_t* Cb, int ldo,
                 int N, int Nround, int Kp, int mode) {
        dim3 grid(Nround / 128, Bsz / 128);
        gemm_bf16_wmma<<<grid, dim3(256), 0, stream>>>(
            A, lda, warena + wOfs, ldb, bias, Cf, ldc, Cb, ldo, N, Kp, mode);
    };

    // 3) fc1: f1 = x @ Wfc1^T + b  -> act0 (bf16)
    G(xbf, KP_F, OF_FC1, KP_F, (const float*)d_in[4],
      nullptr, 0, act0, ACT_LD, Hdim, NR_H, KP_F, 1);

    // 4) GRU layer 1 (fully fused, no fp32 gate scratch): act0=f1, hbf=h1 -> act1=g1
    {
        dim3 grid(NR_H / 128, Bsz / 128);
        gru_fused_wmma<<<grid, dim3(256), 0, stream>>>(
            act0, hbf, warena + OF_GI1, warena + OF_GH1, h1,
            (const float*)d_in[11], (const float*)d_in[12], (const float*)d_in[13],
            (const float*)d_in[14], (const float*)d_in[15], (const float*)d_in[16],
            act1);
    }

    // 5) GRU layer 2: act1=g1, hbf=h2 -> act0=g2
    cvt_pad<<<dim3((ACT_LD + 255) / 256, Bsz), dim3(256), 0, stream>>>(h2, Hdim, hbf, ACT_LD);
    {
        dim3 grid(NR_H / 128, Bsz / 128);
        gru_fused_wmma<<<grid, dim3(256), 0, stream>>>(
            act1, hbf, warena + OF_GI2, warena + OF_GH2, h2,
            (const float*)d_in[23], (const float*)d_in[24], (const float*)d_in[25],
            (const float*)d_in[26], (const float*)d_in[27], (const float*)d_in[28],
            act0);
    }

    // 6) fc2: relu -> act1 ; fc3: relu -> act1->act0... ; fc4: sigmoid -> out
    G(act0, ACT_LD, OF_FC2, KP_H,  (const float*)d_in[30],
      nullptr, 0, act1, ACT_LD, FFdim, NR_FF, KP_H, 2);
    G(act1, ACT_LD, OF_FC3, KP_FF, (const float*)d_in[32],
      nullptr, 0, act0, ACT_LD, FFdim, NR_FF, KP_FF, 2);
    G(act0, ACT_LD, OF_FC4, KP_FF, (const float*)d_in[34],
      out, Fdim, nullptr, 0, Fdim, NR_F, KP_FF, 3);
}
